// HazeAttention_84112639525101
// MI455X (gfx1250) — matrix operand order repaired and checked by reading, not compiled, not run on hardware
//
#include <hip/hip_runtime.h>
#include <hip/hip_bf16.h>

// ---------------------------------------------------------------------------
// HazeAttention on gfx1250 (MI455X): bf16 WMMA everywhere, f32 accumulate.
// ---------------------------------------------------------------------------

typedef __attribute__((ext_vector_type(16))) __bf16 v16bf;
typedef __attribute__((ext_vector_type(8)))  float  v8f;

union V16 { v16bf v; unsigned short h[16]; unsigned int u[8]; };
union V8F { v8f v; float f[8]; };

#define B_  2
#define T_  2048
#define E_  768
#define H_  12
#define D_  64
#define HD_ 768
#define M_  (B_ * T_)   // 4096 rows of x / attn-out

__device__ __forceinline__ unsigned short f2bf(float x) {
    unsigned int u = __float_as_uint(x);
    unsigned int r = u + 0x7FFFu + ((u >> 16) & 1u);   // round-to-nearest-even
    return (unsigned short)(r >> 16);
}

// A-fragment (16 rows x 32 K, bf16, row-major buffer, row stride ld):
// lane l: row = row0 + (l&15); half o=l>>4; v16 pair r holds K = 16*(r/4)+8*o+2*(r%4)
__device__ __forceinline__ v16bf load_a_frag(const unsigned short* buf, int ld,
                                             int row0, int kb, int lane) {
    V16 a;
    int o = (lane >> 4) & 1;
    const unsigned short* p = buf + (row0 + (lane & 15)) * ld + kb;
#pragma unroll
    for (int r = 0; r < 8; ++r) {
        int k = 16 * (r >> 2) + 8 * o + 2 * (r & 3);
        a.u[r] = *(const unsigned int*)(p + k);
    }
    return a.v;
}

// B-fragment from an operand stored TRANSPOSED (rows = N-col, cols = K, stride ld):
// lane l: col n = ncol0 + (l&15); half o=l>>4; v16 pair r holds K = kb + 16*(r/4)+8*o+2*(r%4)
__device__ __forceinline__ v16bf load_bT_frag(const unsigned short* buf, int ld,
                                              int ncol0, int kb, int lane) {
    V16 b;
    int o = (lane >> 4) & 1;
    const unsigned short* p = buf + (ncol0 + (lane & 15)) * ld + kb;
#pragma unroll
    for (int r = 0; r < 8; ++r) b.u[r] = *(const unsigned int*)(p + 16 * (r >> 2) + 8 * o + 2 * (r & 3));
    return b.v;
}

__device__ __forceinline__ float redmax16(float v) {
#pragma unroll
    for (int off = 1; off < 16; off <<= 1) v = fmaxf(v, __shfl_xor(v, off));
    return v;
}
__device__ __forceinline__ float redsum16(float v) {
#pragma unroll
    for (int off = 1; off < 16; off <<= 1) v += __shfl_xor(v, off);
    return v;
}

// ---------------------------------------------------------------------------
// Generic WMMA GEMM: C(MxN) = A(MxK) * B(KxN). Block tile 64x64, 4 waves,
// K staged 64 at a time (8 WMMAs per wave per stage).
// ---------------------------------------------------------------------------
template <bool A_BF16, bool SCATTER>
__global__ __launch_bounds__(128) void gemm_wmma(const void* __restrict__ A_,
                                                 const float* __restrict__ Bm,
                                                 void* __restrict__ outp,
                                                 int M, int N, int K, int ldA) {
    __shared__ unsigned short lsA[64 * 64];
    __shared__ unsigned short lsBT[64 * 64];

    const int tid  = threadIdx.x;
    const int wid  = tid >> 5;
    const int lane = tid & 31;
    const int m0   = blockIdx.x * 64;
    const int n0   = blockIdx.y * 64;

    V8F acc[4];
#pragma unroll
    for (int nc = 0; nc < 4; ++nc) acc[nc].v = (v8f){};

    for (int k0 = 0; k0 < K; k0 += 64) {
        // stage A tile (64x64) as bf16, packed b32 LDS writes
#pragma unroll 4
        for (int idx = tid; idx < 2048; idx += 128) {     // uint pair index
            int r = idx >> 5, cp = idx & 31;              // cp in [0,32)
            if (A_BF16) {
                ((unsigned int*)lsA)[idx] =
                    *(const unsigned int*)((const unsigned short*)A_ +
                                           (size_t)(m0 + r) * ldA + k0 + 2 * cp);
            } else {
                const float* src = (const float*)A_ + (size_t)(m0 + r) * ldA + k0 + 2 * cp;
                unsigned int lo = f2bf(src[0]), hi = f2bf(src[1]);
                ((unsigned int*)lsA)[idx] = lo | (hi << 16);
            }
        }
        // stage B tile transposed: lsBT[n_local][k_local], packed along k
#pragma unroll 4
        for (int idx = tid; idx < 2048; idx += 128) {
            int klp = idx >> 6, nl = idx & 63;            // klp in [0,32)
            unsigned int lo = f2bf(Bm[(size_t)(k0 + 2 * klp) * N + n0 + nl]);
            unsigned int hi = f2bf(Bm[(size_t)(k0 + 2 * klp + 1) * N + n0 + nl]);
            ((unsigned int*)lsBT)[nl * 32 + klp] = lo | (hi << 16);
        }
        if (k0 + 64 < K)                                  // gfx1250 global_prefetch_b8
            __builtin_prefetch(&Bm[(size_t)(k0 + 64) * N + n0 + tid], 0, 1);
        __syncthreads();

#pragma unroll
        for (int kc = 0; kc < 2; ++kc) {
            v16bf a = load_a_frag(lsA, 64, wid * 16, kc * 32, lane);
#pragma unroll
            for (int nc = 0; nc < 4; ++nc) {
                v16bf b = load_bT_frag(lsBT, 64, nc * 16, kc * 32, lane);
                acc[nc].v = __builtin_amdgcn_wmma_f32_16x16x32_bf16(
                    false, a, false, b, (short)0, acc[nc].v, false, false);
            }
        }
        __syncthreads();
    }

    const int o  = (lane >> 4) & 1;
    const int nn = lane & 15;
#pragma unroll
    for (int nc = 0; nc < 4; ++nc) {
#pragma unroll
        for (int i = 0; i < 8; ++i) {
            int gm = m0 + wid * 16 + 8 * o + i;
            int gc = n0 + nc * 16 + nn;
            if (SCATTER) {
                int b = gm / T_, t = gm % T_;
                int h = gc / D_, d = gc % D_;
                ((unsigned short*)outp)[(((size_t)(b * H_ + h) * T_) + t) * D_ + d] =
                    f2bf(acc[nc].f[i]);
            } else {
                ((float*)outp)[(size_t)gm * N + gc] = acc[nc].f[i];
            }
        }
    }
}

// ---------------------------------------------------------------------------
// r[b,h,t] = scale * sum_e x[b,t,e] * wr[h,e,t]    (0.3% of FLOPs -> VALU)
// ---------------------------------------------------------------------------
__global__ __launch_bounds__(256) void rrp_kernel(const float* __restrict__ x,
                                                  const float* __restrict__ wr,
                                                  float* __restrict__ r, float scale) {
    const int ntb = T_ / 256;
    int bx = blockIdx.x;
    int b  = bx / (H_ * ntb);
    int rm = bx % (H_ * ntb);
    int h  = rm / ntb;
    int t  = (rm % ntb) * 256 + threadIdx.x;

    const float* xr = x + ((size_t)b * T_ + t) * E_;
    const float* wc = wr + (size_t)h * E_ * T_ + t;       // stride T_ along e
    float acc = 0.f;
#pragma unroll 4
    for (int e = 0; e < E_; ++e) acc += xr[e] * wc[(size_t)e * T_];
    r[((size_t)b * H_ + h) * T_ + t] = acc * scale;
}

// ---------------------------------------------------------------------------
// Fused causal attention, two online-softmax streams (content + rr) mixed by
// sigmoid(alpha[h]). 4 waves x 16 queries = 64 queries/block; key tiles of 64.
// Wave-uniform tile classes: skipped / full (no masking, rr stats shared
// across rows) / one diagonal tile (masked path).
// ---------------------------------------------------------------------------
__global__ __launch_bounds__(128) void attn_kernel(const unsigned short* __restrict__ Q,
                                                   const unsigned short* __restrict__ Kb,
                                                   const unsigned short* __restrict__ Vb,
                                                   const float* __restrict__ r,
                                                   const float* __restrict__ alpha,
                                                   unsigned short* __restrict__ outA,
                                                   float scale) {
    __shared__ unsigned short lsK[64 * 64];      // [key][d]
    __shared__ unsigned short lsVT[64 * 64];     // [d][key]
    __shared__ unsigned short lsPC[4][16 * 64];  // per-wave P (content)
    __shared__ unsigned short lsPR[4][16 * 64];  // per-wave P (rr)

    const int nQ  = T_ / 64;
    int bx = blockIdx.x;
    int b  = bx / (H_ * nQ);
    int rm = bx % (H_ * nQ);
    int h  = rm / nQ;
    int q0 = (rm % nQ) * 64;

    const int tid  = threadIdx.x;
    const int wid  = tid >> 5;
    const int lane = tid & 31;
    const int o    = (lane >> 4) & 1;
    const int nn   = lane & 15;
    const int wq0  = q0 + wid * 16;              // wave-uniform

    const unsigned short* Qp = Q + (size_t)(b * H_ + h) * T_ * D_;
    const unsigned short* Kp = Kb + (size_t)(b * H_ + h) * T_ * D_;
    const unsigned short* Vp = Vb + (size_t)(b * H_ + h) * T_ * D_;
    const float*          rb = r + (size_t)(b * H_ + h) * T_;

    v16bf aq0 = load_a_frag(Qp, D_, wq0, 0, lane);
    v16bf aq1 = load_a_frag(Qp, D_, wq0, 32, lane);

    V8F oc[4], orr[4];
    float mC[8], mR[8], lC[8], lR[8];
#pragma unroll
    for (int nc = 0; nc < 4; ++nc) { oc[nc].v = (v8f){}; orr[nc].v = (v8f){}; }
#pragma unroll
    for (int i = 0; i < 8; ++i) { mC[i] = -1e30f; mR[i] = -1e30f; lC[i] = 0.f; lR[i] = 0.f; }

    const int jend = q0 + 64;
    for (int j0 = 0; j0 < jend; j0 += 64) {
        // ---- stage K (straight copy) and V (transposed), packed b32 ----
        {
            const unsigned int* Ksrc = (const unsigned int*)(Kp + (size_t)j0 * 64);
            unsigned int* Kdst = (unsigned int*)lsK;
#pragma unroll 4
            for (int idx = tid; idx < 2048; idx += 128) Kdst[idx] = Ksrc[idx];
#pragma unroll 4
            for (int idx = tid; idx < 2048; idx += 128) {
                int kp2 = idx >> 6, d = idx & 63;          // kp2 in [0,32)
                unsigned int lo = Vp[(size_t)(j0 + 2 * kp2) * 64 + d];
                unsigned int hi = Vp[(size_t)(j0 + 2 * kp2 + 1) * 64 + d];
                ((unsigned int*)lsVT)[d * 32 + kp2] = lo | (hi << 16);
            }
        }
        __syncthreads();

        if (j0 <= wq0 + 15) {                   // wave-uniform: tile not fully masked
            // ---- S = Q K^T : 4 N-chunks x 2 K-chunks ----
            V8F s[4];
#pragma unroll
            for (int nc = 0; nc < 4; ++nc) {
                s[nc].v = (v8f){};
                v16bf b0 = load_bT_frag(lsK, 64, nc * 16, 0, lane);
                s[nc].v = __builtin_amdgcn_wmma_f32_16x16x32_bf16(
                    false, aq0, false, b0, (short)0, s[nc].v, false, false);
                v16bf b1 = load_bT_frag(lsK, 64, nc * 16, 32, lane);
                s[nc].v = __builtin_amdgcn_wmma_f32_16x16x32_bf16(
                    false, aq1, false, b1, (short)0, s[nc].v, false, false);
            }

            float rj[4];
#pragma unroll
            for (int f = 0; f < 4; ++f) rj[f] = rb[j0 + 16 * f + nn];

            unsigned short* PwC = &lsPC[wid][0];
            unsigned short* PwR = &lsPR[wid][0];
            const bool full = (j0 + 63 <= wq0);  // wave-uniform

            if (full) {
                // rr tile stats are row-independent: one reduction for all rows
                float mtr = fmaxf(fmaxf(rj[0], rj[1]), fmaxf(rj[2], rj[3]));
                mtr = redmax16(mtr);
                float pb[4], Sr = 0.f;
#pragma unroll
                for (int f = 0; f < 4; ++f) { pb[f] = __expf(rj[f] - mtr); Sr += pb[f]; }
                Sr = redsum16(Sr);

#pragma unroll
                for (int f = 0; f < 4; ++f)
#pragma unroll
                    for (int i = 0; i < 8; ++i) s[f].f[i] *= scale;

#pragma unroll
                for (int i = 0; i < 8; ++i) {
                    // content stream
                    float mc = fmaxf(fmaxf(s[0].f[i], s[1].f[i]),
                                     fmaxf(s[2].f[i], s[3].f[i]));
                    mc = redmax16(mc);
                    float mnC = fmaxf(mC[i], mc);
                    float scC = __expf(mC[i] - mnC);
                    float sumC = 0.f;
#pragma unroll
                    for (int f = 0; f < 4; ++f) {
                        float p = __expf(s[f].f[i] - mnC);
                        sumC += p;
                        PwC[(8 * o + i) * 64 + 16 * f + nn] = f2bf(p);
                    }
                    sumC = redsum16(sumC);
                    lC[i] = lC[i] * scC + sumC; mC[i] = mnC;
                    // rr stream: per-row just one exp + fma
                    float mnR = fmaxf(mR[i], mtr);
                    float scR = __expf(mR[i] - mnR);
                    float fac = __expf(mtr - mnR);
                    lR[i] = lR[i] * scR + Sr * fac; mR[i] = mnR;
#pragma unroll
                    for (int f = 0; f < 4; ++f)
                        PwR[(8 * o + i) * 64 + 16 * f + nn] = f2bf(pb[f] * fac);
#pragma unroll
                    for (int nc = 0; nc < 4; ++nc) { oc[nc].f[i] *= scC; orr[nc].f[i] *= scR; }
                }
            } else {
                // diagonal tile: masked path (runs once per wave)
#pragma unroll
                for (int f = 0; f < 4; ++f) {
                    int j = j0 + 16 * f + nn;
#pragma unroll
                    for (int i = 0; i < 8; ++i) {
                        int q = wq0 + 8 * o + i;
                        s[f].f[i] = (j <= q) ? s[f].f[i] * scale : -1e30f;
                    }
                }
#pragma unroll
                for (int i = 0; i < 8; ++i) {
                    int q = wq0 + 8 * o + i;
                    float mc = fmaxf(fmaxf(s[0].f[i], s[1].f[i]),
                                     fmaxf(s[2].f[i], s[3].f[i]));
                    mc = redmax16(mc);
                    float mr = -1e30f;
#pragma unroll
                    for (int f = 0; f < 4; ++f) {
                        int j = j0 + 16 * f + nn;
                        if (j <= q) mr = fmaxf(mr, rj[f]);
                    }
                    mr = redmax16(mr);
                    float mnC = fmaxf(mC[i], mc), mnR = fmaxf(mR[i], mr);
                    float scC = __expf(mC[i] - mnC), scR = __expf(mR[i] - mnR);
                    float sumC = 0.f, sumR = 0.f;
#pragma unroll
                    for (int f = 0; f < 4; ++f) {
                        int j = j0 + 16 * f + nn;
                        bool keep = (j <= q);
                        float pc = keep ? __expf(s[f].f[i] - mnC) : 0.f;
                        float pq = keep ? __expf(rj[f] - mnR) : 0.f;
                        sumC += pc; sumR += pq;
                        PwC[(8 * o + i) * 64 + 16 * f + nn] = f2bf(pc);
                        PwR[(8 * o + i) * 64 + 16 * f + nn] = f2bf(pq);
                    }
                    sumC = redsum16(sumC);
                    sumR = redsum16(sumR);
                    lC[i] = lC[i] * scC + sumC; lR[i] = lR[i] * scR + sumR;
                    mC[i] = mnC; mR[i] = mnR;
#pragma unroll
                    for (int nc = 0; nc < 4; ++nc) { oc[nc].f[i] *= scC; orr[nc].f[i] *= scR; }
                }
            }

            // ---- PV for both streams: P(16x64) * V(64x64) ----
#pragma unroll
            for (int pa = 0; pa < 2; ++pa) {
                v16bf apc = load_a_frag(PwC, 64, 0, pa * 32, lane);
#pragma unroll
                for (int nc = 0; nc < 4; ++nc) {
                    v16bf bv = load_bT_frag(lsVT, 64, nc * 16, pa * 32, lane);
                    oc[nc].v = __builtin_amdgcn_wmma_f32_16x16x32_bf16(
                        false, apc, false, bv, (short)0, oc[nc].v, false, false);
                }
                v16bf apr = load_a_frag(PwR, 64, 0, pa * 32, lane);
#pragma unroll
                for (int nc = 0; nc < 4; ++nc) {
                    v16bf bv = load_bT_frag(lsVT, 64, nc * 16, pa * 32, lane);
                    orr[nc].v = __builtin_amdgcn_wmma_f32_16x16x32_bf16(
                        false, apr, false, bv, (short)0, orr[nc].v, false, false);
                }
            }
        }
        __syncthreads();
    }

    float aH = 1.f / (1.f + __expf(-alpha[h]));
#pragma unroll
    for (int nc = 0; nc < 4; ++nc) {
#pragma unroll
        for (int i = 0; i < 8; ++i) {
            int q = wq0 + 8 * o + i;
            int d = nc * 16 + nn;
            float val = aH * (orr[nc].f[i] / lR[i]) + (1.f - aH) * (oc[nc].f[i] / lC[i]);
            outA[((size_t)b * T_ + q) * HD_ + h * D_ + d] = f2bf(val);
        }
    }
}

// ---------------------------------------------------------------------------
extern "C" void kernel_launch(void* const* d_in, const int* in_sizes, int n_in,
                              void* d_out, int out_size, void* d_ws, size_t ws_size,
                              hipStream_t stream) {
    const float* x     = (const float*)d_in[0];
    const float* wq    = (const float*)d_in[1];
    const float* wk    = (const float*)d_in[2];
    const float* wv    = (const float*)d_in[3];
    const float* wr    = (const float*)d_in[4];
    const float* alpha = (const float*)d_in[5];
    const float* wo    = (const float*)d_in[6];

    const size_t nBHTD = (size_t)B_ * H_ * T_ * D_;     // 3,145,728 elems
    char* ws = (char*)d_ws;
    unsigned short* Qw = (unsigned short*)(ws);
    unsigned short* Kw = (unsigned short*)(ws + nBHTD * 2);
    unsigned short* Vw = (unsigned short*)(ws + nBHTD * 4);
    unsigned short* Aw = (unsigned short*)(ws + nBHTD * 6);          // [B,T,H*D] bf16
    float*          rw = (float*)(ws + nBHTD * 8);                   // [B,H,T] f32

    const float scale = 1.0f / 8.0f;  // 1/sqrt(D=64)

    dim3 gGemm(M_ / 64, HD_ / 64);   // (64, 12)
    gemm_wmma<false, true><<<gGemm, 128, 0, stream>>>(x, wq, Qw, M_, HD_, E_, E_);
    gemm_wmma<false, true><<<gGemm, 128, 0, stream>>>(x, wk, Kw, M_, HD_, E_, E_);
    gemm_wmma<false, true><<<gGemm, 128, 0, stream>>>(x, wv, Vw, M_, HD_, E_, E_);

    rrp_kernel<<<B_ * H_ * (T_ / 256), 256, 0, stream>>>(x, wr, rw, scale);

    attn_kernel<<<B_ * H_ * (T_ / 64), 128, 0, stream>>>(Qw, Kw, Vw, rw, alpha, Aw, scale);

    gemm_wmma<true, false><<<gGemm, 128, 0, stream>>>(Aw, wo, d_out, M_, E_, HD_, HD_);
}
